// GRUModel_77927886619307
// MI455X (gfx1250) — matrix operand order repaired and checked by reading, not compiled, not run on hardware
//
#include <hip/hip_runtime.h>
#include <hip/hip_bf16.h>
#include <stdint.h>

// ---------------------------------------------------------------------------
// Model dims
// ---------------------------------------------------------------------------
#define BB   64
#define TT   2048
#define DD   80
#define DP   96          // D padded to 3 chunks of 32 for WMMA K
#define HH   128
#define H2   256
#define GG   384         // 3*H
#define OO   4
#define MROWS (BB * TT)  // 131072
#define MTILES (MROWS / 16)  // 8192

typedef __attribute__((ext_vector_type(16))) _Float16 v16h;
typedef __attribute__((ext_vector_type(8)))  _Float16 v8h;
typedef __attribute__((ext_vector_type(8)))  float    v8f;
typedef int v4i_t __attribute__((vector_size(16)));   // matches async builtin

union FragU { v16h v; uint4 q[2]; };

#if defined(__HIP_DEVICE_COMPILE__)
#if __has_builtin(__builtin_amdgcn_global_load_async_to_lds_b128)
#define ASYNC_LDS 1
#endif
#endif

// ---------------------------------------------------------------------------
// WMMA fragment helpers (layouts per CDNA5 ISA 7.12.2, wave32)
// A 16x32 f16:  lane m = lane&15 ; halves j0..7 -> K = kc*32 + (lane>>4)*8 + j
//               halves j8..15 -> K = kc*32 + 16 + (lane>>4)*8 + (j-8)
// B 32x16 f16 from row-major (N,K) weights: n = lane&15 ;
//               halves j0..7 -> K = kc*32 + (lane>>4)*8 + j, j8..15 -> K = kc*32 + 16 + (lane>>4)*8 + (j-8)
// C/D 16x16 f32: vgpr r, lane L: M = (L>>4)*8 + r, N = L&15
// ---------------------------------------------------------------------------
__device__ __forceinline__ v16h frag_a(const _Float16* tile, int pitch, int kc) {
    const int lane = threadIdx.x & 31;
    const _Float16* p = tile + (lane & 15) * pitch + kc * 32 + ((lane >> 4) << 3);
    FragU u;
    u.q[0] = *(const uint4*)(p);
    u.q[1] = *(const uint4*)(p + 16);
    return u.v;
}

__device__ __forceinline__ v16h frag_b(const _Float16* w, int pitch, int kc) {
    const int lane = threadIdx.x & 31;
    const _Float16* p = w + (lane & 15) * pitch + kc * 32 + ((lane >> 4) << 3);
    FragU u;
    u.q[0] = *(const uint4*)(p);
    u.q[1] = *(const uint4*)(p + 16);
    return u.v;
}

__device__ __forceinline__ v8f wmma_f16(v16h a, v16h b, v8f c) {
    return __builtin_amdgcn_wmma_f32_16x16x32_f16(false, a, false, b, (short)0, c,
                                                  false, false);
}

__device__ __forceinline__ v8f zero8() { v8f z = {}; return z; }

__device__ __forceinline__ float sigmf(float x) {
    return 1.0f / (1.0f + __expf(-x));
}

// ---------------------------------------------------------------------------
// Pack all weights to fp16 layouts.
//  wfront: [10][256][96]  mats: 0=conv1 tap0, 1..3=conv3 taps, 4..8=conv5 taps,
//                               9=fc1    (K zero-padded 80->96)
//  wih16:  [L][2][384][256]  straight convert of gru_wih
//  fc16:   [16][256]  fc_w rows 0..3, rows 4..15 zero
// ---------------------------------------------------------------------------
__global__ __launch_bounds__(256) void pack_kernel(
    const float* __restrict__ c1w, const float* __restrict__ c3w,
    const float* __restrict__ c5w, const float* __restrict__ f1w,
    const float* __restrict__ wih, const float* __restrict__ fcw,
    _Float16* __restrict__ wfront, _Float16* __restrict__ wih16,
    _Float16* __restrict__ fc16) {
    const int NF = 10 * H2 * DP;
    const int NW = 2 * 2 * GG * H2;
    const int NC = 16 * H2;
    int idx = blockIdx.x * 256 + threadIdx.x;
    if (idx < NF) {
        int mat = idx / (H2 * DP);
        int rem = idx - mat * H2 * DP;
        int n = rem / DP, k = rem - n * DP;
        float v = 0.0f;
        if (k < DD) {
            if (mat == 0)      v = c1w[n * DD + k];                   // (256,80,1)
            else if (mat <= 3) v = c3w[(n * DD + k) * 3 + (mat - 1)]; // (256,80,3)
            else if (mat <= 8) v = c5w[(n * DD + k) * 5 + (mat - 4)]; // (256,80,5)
            else               v = f1w[n * DD + k];                   // (256,80)
        }
        wfront[idx] = (_Float16)v;
    } else if (idx < NF + NW) {
        int j = idx - NF;
        wih16[j] = (_Float16)wih[j];
    } else if (idx < NF + NW + NC) {
        int j = idx - NF - NW;
        int n = j / H2, k = j - n * H2;
        fc16[j] = (_Float16)(n < OO ? fcw[n * H2 + k] : 0.0f);
    }
}

// ---------------------------------------------------------------------------
// Frontend: h = relu(conv1)+relu(conv3)+relu(conv5), xfc1 = x@fc1^T + b
// One block = one 16-row m-tile (16 consecutive t within one batch b).
// x_ext in LDS covers t0-2 .. t0+17 (zero padded), K padded to 96.
// 8 waves x 2 n-tiles each cover all 256 output channels.
// ---------------------------------------------------------------------------
__global__ __launch_bounds__(256) void frontend_kernel(
    const float* __restrict__ x, const _Float16* __restrict__ wf,
    const float* __restrict__ b1v, const float* __restrict__ b3v,
    const float* __restrict__ b5v, const float* __restrict__ bfv,
    _Float16* __restrict__ h0, _Float16* __restrict__ xfc1) {
    __shared__ __align__(16) _Float16 xe[20 * DP];
    const int mtile = blockIdx.x;
    const int b  = mtile >> 7;             // 128 tiles per batch row
    const int t0 = (mtile & 127) << 4;
    const int tid = threadIdx.x;
    for (int idx = tid; idx < 20 * DP; idx += 256) {
        int r = idx / DP, d = idx - r * DP;
        int t = t0 + r - 2;
        float v = 0.0f;
        if (t >= 0 && t < TT && d < DD)
            v = x[((size_t)b * TT + t) * DD + d];
        xe[idx] = (_Float16)v;
    }
    __syncthreads();

    const int wave = tid >> 5, lane = tid & 31;
    const int Nn = lane & 15, hi = lane >> 4;
    const size_t row0 = (size_t)mtile * 16;

    for (int nt = wave * 2; nt < wave * 2 + 2; ++nt) {
        const int col = nt * 16 + Nn;
        const float bc1 = b1v[col], bc3 = b3v[col], bc5 = b5v[col], bcf = bfv[col];
        v8f hsum = zero8();
        v8f acc;
        // conv1 (tap shift 0)
        acc = zero8();
        {
            const _Float16* wb = wf + ((size_t)0 * H2 + nt * 16) * DP;
            for (int kc = 0; kc < 3; ++kc)
                acc = wmma_f16(frag_a(xe + 2 * DP, DP, kc), frag_b(wb, DP, kc), acc);
        }
        for (int r = 0; r < 8; ++r) {
            float v = acc[r] + bc1; hsum[r] += v > 0.0f ? v : 0.0f;
        }
        // conv3 (shifts -1..1, mats 1..3)
        acc = zero8();
        for (int tap = 0; tap < 3; ++tap) {
            const _Float16* wb = wf + ((size_t)(1 + tap) * H2 + nt * 16) * DP;
            const _Float16* at = xe + (2 + tap - 1) * DP;
            for (int kc = 0; kc < 3; ++kc)
                acc = wmma_f16(frag_a(at, DP, kc), frag_b(wb, DP, kc), acc);
        }
        for (int r = 0; r < 8; ++r) {
            float v = acc[r] + bc3; hsum[r] += v > 0.0f ? v : 0.0f;
        }
        // conv5 (shifts -2..2, mats 4..8)
        acc = zero8();
        for (int tap = 0; tap < 5; ++tap) {
            const _Float16* wb = wf + ((size_t)(4 + tap) * H2 + nt * 16) * DP;
            const _Float16* at = xe + (2 + tap - 2) * DP;
            for (int kc = 0; kc < 3; ++kc)
                acc = wmma_f16(frag_a(at, DP, kc), frag_b(wb, DP, kc), acc);
        }
        for (int r = 0; r < 8; ++r) {
            float v = acc[r] + bc5; hsum[r] += v > 0.0f ? v : 0.0f;
        }
        // fc1 (shift 0, mat 9)
        v8f accf = zero8();
        {
            const _Float16* wb = wf + ((size_t)9 * H2 + nt * 16) * DP;
            for (int kc = 0; kc < 3; ++kc)
                accf = wmma_f16(frag_a(xe + 2 * DP, DP, kc), frag_b(wb, DP, kc), accf);
        }
        for (int r = 0; r < 8; ++r) {
            size_t row = row0 + hi * 8 + r;
            h0[row * H2 + col]   = (_Float16)hsum[r];
            xfc1[row * H2 + col] = (_Float16)(accf[r] + bcf);
        }
    }
}

// ---------------------------------------------------------------------------
// xg = h_in @ wih^T + bih  for one layer, both directions.
// M = 131072, N = 768 (dir-major), K = 256.  One block = one m-tile,
// 8 waves x 6 n-tiles x 8 k-chunks = 48 WMMA per wave.
// A tile staged via GLOBAL_LOAD_ASYNC_TO_LDS_B128 (ASYNCcnt) when available.
// Output layout xg[dir][b][t][384] (fp16) so the scan can walk time directly.
// ---------------------------------------------------------------------------
__global__ __launch_bounds__(256) void xgemm_kernel(
    const _Float16* __restrict__ hin, const _Float16* __restrict__ wih,
    const float* __restrict__ bih, _Float16* __restrict__ xg) {
    __shared__ __align__(16) _Float16 As[16 * H2];
    const int tid = threadIdx.x;
    const size_t row0 = (size_t)blockIdx.x * 16;
#ifdef ASYNC_LDS
    {
        const _Float16* src = hin + row0 * H2;
        // 16x256 halves = 512 x 16B chunks; 2 async B128 transfers per thread
        for (int c = tid; c < 512; c += 256) {
            __builtin_amdgcn_global_load_async_to_lds_b128(
                (__attribute__((address_space(1))) v4i_t*)(src + c * 8),
                (__attribute__((address_space(3))) v4i_t*)(As + c * 8), 0, 0);
        }
#if __has_builtin(__builtin_amdgcn_s_wait_asynccnt)
        __builtin_amdgcn_s_wait_asynccnt(0);
#else
        asm volatile("s_wait_asynccnt 0" ::: "memory");
#endif
    }
#else
    for (int idx = tid; idx < 16 * H2; idx += 256)
        As[idx] = hin[row0 * H2 + idx];
#endif
    __syncthreads();

    const int wave = tid >> 5, lane = tid & 31;
    const int Nn = lane & 15, hi = lane >> 4;
    for (int nt = wave * 6; nt < wave * 6 + 6; ++nt) {
        const _Float16* wb = wih + (size_t)nt * 16 * H2;
        __builtin_prefetch(wb + 16 * H2, 0, 0);
        v8f acc = zero8();
        for (int kc = 0; kc < 8; ++kc)
            acc = wmma_f16(frag_a(As, H2, kc), frag_b(wb, H2, kc), acc);
        const int g768 = nt * 16 + Nn;
        const float bias = bih[g768];
        const int dir = g768 >= GG;
        const int g = g768 - dir * GG;
        for (int r = 0; r < 8; ++r) {
            size_t row = row0 + hi * 8 + r;
            size_t bb = row >> 11, tt = row & (TT - 1);
            xg[(((size_t)dir * BB + bb) * TT + tt) * GG + g] =
                (_Float16)(acc[r] + bias);
        }
    }
}

// ---------------------------------------------------------------------------
// Persistent GRU scan, one layer, grid = (4 batch-tiles, 2 dirs).
// whh (384x128 fp16, 96KB) and h live in LDS; per step:
//   - xg/xfc1 rows for this step are prefetched before the matmul (they do
//     not depend on h), bhh is hoisted out of the time loop entirely
//   - phase1: hg = h @ whh^T ; A fragments hoisted across the 3 n-tiles,
//     K accumulation split into two independent 2-deep WMMA chains
//   - phase2: gates + state update + stream output (+xfc1 on final layer)
// ---------------------------------------------------------------------------
__global__ __launch_bounds__(256) void gru_scan_kernel(
    const _Float16* __restrict__ xg, const float* __restrict__ whh,
    const float* __restrict__ bhh, const _Float16* __restrict__ xfc1,
    _Float16* __restrict__ hout16, float* __restrict__ out_gru,
    int final_layer) {
    __shared__ __align__(16) _Float16 wh[GG * HH];     // 96 KB
    __shared__ __align__(16) _Float16 h16[16 * HH];    // 4 KB
    __shared__ float hf[16 * HH];                      // 8 KB
    __shared__ float hg[16 * 392];                     // ~25 KB (pitch 392)
    const int b0  = blockIdx.x * 16;
    const int dir = blockIdx.y;
    const int tid = threadIdx.x;

    const float* wsrc = whh + (size_t)dir * GG * HH;
    for (int idx = tid; idx < GG * HH; idx += 256) wh[idx] = (_Float16)wsrc[idx];
    for (int idx = tid; idx < 16 * HH; idx += 256) {
        h16[idx] = (_Float16)0.0f;
        hf[idx] = 0.0f;
    }
    __syncthreads();

    const int wave = tid >> 5, lane = tid & 31;
    const int Nn = lane & 15, hi = lane >> 4;
    const int em = tid >> 4;          // row 0..15 for elementwise phase
    const int ej0 = (tid & 15) * 8;   // 8 consecutive hidden units

    // loop-invariant recurrent biases for this thread's 8 hidden units
    const float* bh = bhh + dir * GG;
    float bhr[8], bhz[8], bhn[8];
    for (int i = 0; i < 8; ++i) {
        bhr[i] = bh[ej0 + i];
        bhz[i] = bh[HH + ej0 + i];
        bhn[i] = bh[2 * HH + ej0 + i];
    }

    const size_t xrow_base = ((size_t)dir * BB + (b0 + em)) * TT;

    for (int s = 0; s < TT; ++s) {
        const int t = dir ? (TT - 1 - s) : s;
        // --- prefetch this step's input-projection rows (independent of h) ---
        const _Float16* xrow = xg + (xrow_base + t) * GG;
        const v8h xrv = *(const v8h*)(xrow + ej0);
        const v8h xzv = *(const v8h*)(xrow + HH + ej0);
        const v8h xnv = *(const v8h*)(xrow + 2 * HH + ej0);
        const size_t obase = ((size_t)(b0 + em) * TT + t) * H2 + dir * HH + ej0;
        v8h xfv = {};
        if (final_layer) xfv = *(const v8h*)(xfc1 + obase);
        // --- phase 1: recurrent matmul hg = h @ whh^T ---
        {
            const v16h a0 = frag_a(h16, HH, 0), a1 = frag_a(h16, HH, 1);
            const v16h a2 = frag_a(h16, HH, 2), a3 = frag_a(h16, HH, 3);
            for (int nt = wave * 3; nt < wave * 3 + 3; ++nt) {
                const _Float16* wb = wh + nt * 16 * HH;
                v8f e = wmma_f16(a0, frag_b(wb, HH, 0), zero8());
                v8f o = wmma_f16(a2, frag_b(wb, HH, 2), zero8());
                e = wmma_f16(a1, frag_b(wb, HH, 1), e);
                o = wmma_f16(a3, frag_b(wb, HH, 3), o);
                const v8f acc = e + o;
                for (int r = 0; r < 8; ++r)
                    hg[(hi * 8 + r) * 392 + nt * 16 + Nn] = acc[r];
            }
        }
        __syncthreads();
        // --- phase 2: gates + update ---
        for (int i = 0; i < 8; ++i) {
            const int j = ej0 + i;
            const float hgr = hg[em * 392 + j] + bhr[i];
            const float hgz = hg[em * 392 + HH + j] + bhz[i];
            const float hgn = hg[em * 392 + 2 * HH + j] + bhn[i];
            const float rg = sigmf((float)xrv[i] + hgr);
            const float zg = sigmf((float)xzv[i] + hgz);
            const float ng = tanhf((float)xnv[i] + rg * hgn);
            const float hp = hf[em * HH + j];
            const float hn = (1.0f - zg) * ng + zg * hp;
            hf[em * HH + j] = hn;
            h16[em * HH + j] = (_Float16)hn;
            if (final_layer)
                out_gru[obase + i] = hn + (float)xfv[i];
            else
                hout16[obase + i] = (_Float16)hn;
        }
        __syncthreads();
    }
}

// ---------------------------------------------------------------------------
// out = output_gru @ fc_w^T + fc_b  (N=4 zero-padded to 16).
// One wave per m-tile, 8 tiles per block, 1024 blocks.
// ---------------------------------------------------------------------------
__global__ __launch_bounds__(256) void out_kernel(
    const float* __restrict__ og, const _Float16* __restrict__ fcw,
    const float* __restrict__ fcb, float* __restrict__ out) {
    __shared__ __align__(16) _Float16 As[8][16 * H2];  // 64 KB
    const int wave = threadIdx.x >> 5, lane = threadIdx.x & 31;
    const int mtile = blockIdx.x * 8 + wave;
    const size_t row0 = (size_t)mtile * 16;
    for (int idx = lane; idx < 16 * H2; idx += 32)
        As[wave][idx] = (_Float16)og[row0 * H2 + idx];
    __syncthreads();
    const int Nn = lane & 15, hi = lane >> 4;
    v8f acc = zero8();
    for (int kc = 0; kc < 8; ++kc)
        acc = wmma_f16(frag_a(As[wave], H2, kc), frag_b(fcw, H2, kc), acc);
    if (Nn < OO) {
        const float bias = fcb[Nn];
        for (int r = 0; r < 8; ++r)
            out[(row0 + hi * 8 + r) * OO + Nn] = acc[r] + bias;
    }
}

// ---------------------------------------------------------------------------
// Host launcher
// ---------------------------------------------------------------------------
extern "C" void kernel_launch(void* const* d_in, const int* in_sizes, int n_in,
                              void* d_out, int out_size, void* d_ws,
                              size_t ws_size, hipStream_t stream) {
    (void)in_sizes; (void)n_in; (void)out_size; (void)ws_size;
    const float* x      = (const float*)d_in[0];
    const float* c1w    = (const float*)d_in[1];
    const float* c1b    = (const float*)d_in[2];
    const float* c3w    = (const float*)d_in[3];
    const float* c3b    = (const float*)d_in[4];
    const float* c5w    = (const float*)d_in[5];
    const float* c5b    = (const float*)d_in[6];
    const float* f1w    = (const float*)d_in[7];
    const float* f1b    = (const float*)d_in[8];
    const float* fcw    = (const float*)d_in[9];
    const float* fcb    = (const float*)d_in[10];
    const float* g_wih  = (const float*)d_in[11];
    const float* g_whh  = (const float*)d_in[12];
    const float* g_bih  = (const float*)d_in[13];
    const float* g_bhh  = (const float*)d_in[14];

    uint8_t* ws = (uint8_t*)d_ws;
    size_t off = 0;
    auto alloc = [&](size_t bytes) -> void* {
        void* p = ws + off;
        off = (off + bytes + 255) & ~(size_t)255;
        return p;
    };
    _Float16* wfront  = (_Float16*)alloc((size_t)10 * H2 * DP * 2);
    _Float16* wih16   = (_Float16*)alloc((size_t)2 * 2 * GG * H2 * 2);
    _Float16* fc16    = (_Float16*)alloc((size_t)16 * H2 * 2);
    _Float16* h0_16   = (_Float16*)alloc((size_t)MROWS * H2 * 2);
    _Float16* h1_16   = (_Float16*)alloc((size_t)MROWS * H2 * 2);
    _Float16* xfc1_16 = (_Float16*)alloc((size_t)MROWS * H2 * 2);
    _Float16* xg16    = (_Float16*)alloc((size_t)2 * MROWS * GG * 2);

    float* out_gru = (float*)d_out;                       // [B*T][256]
    float* out_fc  = (float*)d_out + (size_t)MROWS * H2;  // [B*T][4]

    const int npack = 10 * H2 * DP + 2 * 2 * GG * H2 + 16 * H2;
    pack_kernel<<<(npack + 255) / 256, 256, 0, stream>>>(
        c1w, c3w, c5w, f1w, g_wih, fcw, wfront, wih16, fc16);

    frontend_kernel<<<MTILES, 256, 0, stream>>>(x, wfront, c1b, c3b, c5b, f1b,
                                                h0_16, xfc1_16);

    for (int layer = 0; layer < 2; ++layer) {
        const _Float16* hin = (layer == 0) ? h0_16 : h1_16;
        xgemm_kernel<<<MTILES, 256, 0, stream>>>(
            hin, wih16 + (size_t)layer * 2 * GG * H2, g_bih + layer * 2 * GG,
            xg16);
        gru_scan_kernel<<<dim3(4, 2), 256, 0, stream>>>(
            xg16, g_whh + (size_t)layer * 2 * GG * HH, g_bhh + layer * 2 * GG,
            xfc1_16, h1_16, out_gru, layer == 1);
    }

    out_kernel<<<MTILES / 8, 256, 0, stream>>>(out_gru, fc16, fcb, out_fc);
}
